// EGNNRegressionHead_52149492908466
// MI455X (gfx1250) — matrix operand order repaired and checked by reading, not compiled, not run on hardware
//
#include <hip/hip_runtime.h>
#include <hip/hip_bf16.h>

typedef __attribute__((ext_vector_type(16))) _Float16 v16h;
typedef __attribute__((ext_vector_type(8)))  float    v8f;

static constexpr int kN       = 50000;
static constexpr int kE       = 800000;
static constexpr int kFIn     = 32;
static constexpr int kH       = 96;
static constexpr int kEdgeDim = 16;
static constexpr int kB       = 50;
static constexpr int kMsgIn   = 2 * kH + 1 + kEdgeDim; // 209
static constexpr int kMsgKp   = 224;                   // 7 * 32
static constexpr int kNodeIn  = 192;                   // 6 * 32
static constexpr int kCT      = kH / 16;               // 6 col tiles
static constexpr int kNodeTiles = kN / 16;             // 3125 (exact)
static constexpr int kEdgeTiles = kE / 16;             // 50000 (exact)

// ---------------- device helpers ----------------

__device__ __forceinline__ void lds_fence() {
  // intra-wave LDS RAW visibility (cross-lane): wait for DS stores to land.
  asm volatile("s_wait_dscnt 0" ::: "memory");
}

__device__ __forceinline__ void async_fence_all() {
  asm volatile("s_wait_asynccnt 0" ::: "memory");
}

// Async DMA: 16 bytes global -> LDS, tracked by ASYNCcnt (no VGPR round trip).
// LDS byte address = low 32 bits of the flat pointer.
__device__ __forceinline__ void async_b128(_Float16* lds, const _Float16* g) {
  unsigned int loff = (unsigned int)(unsigned long long)lds;
  asm volatile("global_load_async_to_lds_b128 %0, %1, off"
               :: "v"(loff), "v"(g) : "memory");
}

__device__ __forceinline__ v8f zero8() {
  v8f z = {0.f, 0.f, 0.f, 0.f, 0.f, 0.f, 0.f, 0.f};
  return z;
}

__device__ __forceinline__ v8f wmma16(v16h a, v16h b, v8f c) {
  // D = A(16x32 f16) * B(32x16 f16) + C(16x16 f32)
  return __builtin_amdgcn_wmma_f32_16x16x32_f16(false, a, false, b, (short)0, c,
                                                false, false);
}

__device__ __forceinline__ float silu(float x) {
  return x / (1.f + __expf(-x));
}

// A fragment (16x32 f16) from a row-major LDS tile [16][strideH] at K-step ks.
// ISA layout: lane -> M = lane%16; base = (lane/16)*8; halves 0..7 -> K=base..base+7,
// halves 8..15 -> K=16+base..16+base+7 (within the 32-wide K step).
__device__ __forceinline__ v16h load_afrag(const _Float16* tile, int strideH, int ks) {
  int lane = threadIdx.x & 31;
  int row  = lane & 15;
  int base = (lane >> 4) << 3;
  const _Float16* p = tile + row * strideH + (ks << 5);
  union { v16h v; uint4 q[2]; } u;
  u.q[0] = *(const uint4*)(p + base);
  u.q[1] = *(const uint4*)(p + 16 + base);
  return u.v;
}

// B fragment from pre-packed weights: packed[((ct*numK+ks)*32 + lane)*16 + h]
__device__ __forceinline__ v16h load_bfrag(const _Float16* pk, int numK, int ct, int ks) {
  int lane = threadIdx.x & 31;
  const _Float16* p = pk + ((size_t)((ct * numK + ks) << 5) + lane) * 16;
  union { v16h v; uint4 q[2]; } u;
  u.q[0] = *(const uint4*)(p);
  u.q[1] = *(const uint4*)(p + 8);
  return u.v;
}

// ---------------- prep kernels ----------------

// Pack f32 weight W[K x 96] (row-major) into per-lane B-fragment layout, f16,
// K zero-padded to numK*32. ISA B layout: lane -> N = lane%16; base = (lane/16)*8; halves 0..7 -> K=base..base+7, halves 8..15 -> K=16+base..16+base+7.
// mode==1 (edge W1): packed-K order is [h_src|h_dst(0:192) | ea(192:208) | d2(208)],
// while the reference rows are [h|h|d2|ea]; permute accordingly.
__global__ void pack_weights(const float* __restrict__ W, int K, int numK, int mode,
                             _Float16* __restrict__ dst) {
  int idx   = blockIdx.x * blockDim.x + threadIdx.x;
  int total = kCT * numK * 512;
  if (idx >= total) return;
  int h    = idx & 15;
  int lane = (idx >> 4) & 31;
  int rem  = idx >> 9;
  int ks   = rem % numK;
  int ct   = rem / numK;
  int n    = ct * 16 + (lane & 15);
  int k    = ks * 32 + ((h >> 3) << 4) + ((lane >> 4) << 3) + (h & 7);
  if (mode == 1) {
    if (k < 192)       ;            // h_src | h_dst rows unchanged
    else if (k < 208)  k = k + 1;   // edge_attr: ref rows 193..208
    else if (k == 208) k = 192;     // d2: ref row 192
    else               k = -1;      // pad
  }
  float v  = (k >= 0 && k < K) ? W[(size_t)k * kH + n] : 0.f;
  dst[idx] = (_Float16)v;
}

__global__ void zero_kernel(float* __restrict__ p, int n) {
  int i = blockIdx.x * blockDim.x + threadIdx.x;
  if (i < n) p[i] = 0.f;
}

// Per edge: f16 edge_attr block and a 16-half padded block [d2, 0 x 15]
// so the whole edge A-tile gather is aligned async-B128 traffic.
__global__ void edge_pre(const int* __restrict__ ei, const float* __restrict__ pos,
                         const float* __restrict__ ea,
                         _Float16* __restrict__ ea16, _Float16* __restrict__ d2p) {
  int e = blockIdx.x * blockDim.x + threadIdx.x;
  if (e >= kE) return;
  int s = ei[e];
  int d = ei[kE + e];
  float dx = pos[s * 3 + 0] - pos[d * 3 + 0];
  float dy = pos[s * 3 + 1] - pos[d * 3 + 1];
  float dz = pos[s * 3 + 2] - pos[d * 3 + 2];
  float d2 = dx * dx + dy * dy + dz * dz;
#pragma unroll
  for (int i = 0; i < kEdgeDim; i++)
    ea16[(size_t)e * kEdgeDim + i] = (_Float16)ea[(size_t)e * kEdgeDim + i];
  d2p[(size_t)e * 16] = (_Float16)d2;
#pragma unroll
  for (int i = 1; i < 16; i++) d2p[(size_t)e * 16 + i] = (_Float16)0.f;
}

// ---------------- encoder: h = x @ We + be ----------------

#define ENC_WAVES 8
__global__ void encoder_kernel(const float* __restrict__ x,
                               const _Float16* __restrict__ pWe,
                               const float* __restrict__ be,
                               float* __restrict__ h32,
                               _Float16* __restrict__ h16) {
  __shared__ __align__(16) _Float16 sm[ENC_WAVES * 16 * kFIn];
  int wave = threadIdx.x >> 5;
  int lane = threadIdx.x & 31;
  int tile = blockIdx.x * ENC_WAVES + wave;
  if (tile >= kNodeTiles) return;
  _Float16* A = sm + wave * 16 * kFIn;
  {
    int row = lane & 15, half = lane >> 4;
    const float* src = x + (size_t)(tile * 16 + row) * kFIn + half * 16;
    _Float16* dp = A + row * kFIn + half * 16;
#pragma unroll
    for (int i = 0; i < 16; i++) dp[i] = (_Float16)src[i];
  }
  lds_fence();
  v16h a = load_afrag(A, kFIn, 0);
  int n = lane & 15, mb = (lane >> 4) * 8;
#pragma unroll
  for (int ct = 0; ct < kCT; ct++) {
    v16h b = load_bfrag(pWe, 1, ct, 0);
    v8f acc = wmma16(a, b, zero8());
    float bias = be[ct * 16 + n];
#pragma unroll
    for (int v = 0; v < 8; v++) {
      float val = acc[v] + bias;
      size_t o = (size_t)(tile * 16 + mb + v) * kH + ct * 16 + n;
      h32[o] = val;
      h16[o] = (_Float16)val;
    }
  }
}

// ---------------- edge MLP (async double-buffered gather) ----------------
// A-tile columns: [h_src 0:96 | h_dst 96:192 | ea 192:208 | d2 208 | pad]

#define EDGE_WAVES 4
#define TPW 5   // tiles per wave; kEdgeTiles / TPW / EDGE_WAVES = 2500 blocks exact

__device__ __forceinline__ void edge_gather_async(_Float16* buf, int tile,
                                                  const _Float16* h16,
                                                  const _Float16* ea16,
                                                  const _Float16* d2p,
                                                  const int* src, const int* dst) {
  int lane = threadIdx.x & 31;
  int row  = lane & 15;
  int e    = tile * 16 + row;
  if (lane < 16) {
    int s = src[e];
    const _Float16* hp = h16 + (size_t)s * kH;
    _Float16* dp = buf + row * kMsgKp;
#pragma unroll
    for (int i = 0; i < 12; i++) async_b128(dp + i * 8, hp + i * 8);       // 96 halves
#pragma unroll
    for (int i = 0; i < 2; i++)
      async_b128(dp + 192 + i * 8, ea16 + (size_t)e * 16 + i * 8);         // edge_attr
#pragma unroll
    for (int i = 0; i < 2; i++)
      async_b128(dp + 208 + i * 8, d2p + (size_t)e * 16 + i * 8);          // d2 + pad
  } else {
    int d = dst[e];
    const _Float16* hp = h16 + (size_t)d * kH;
    _Float16* dp = buf + row * kMsgKp + kH;
#pragma unroll
    for (int i = 0; i < 12; i++) async_b128(dp + i * 8, hp + i * 8);
  }
  // 12 + 2 + 2 + 12 = 28 async instructions per tile (wave-wide)
}

__global__ void edge_mlp(const _Float16* __restrict__ h16,
                         const _Float16* __restrict__ ea16,
                         const _Float16* __restrict__ d2p,
                         const int* __restrict__ ei,
                         const _Float16* __restrict__ pW1, const float* __restrict__ b1,
                         const _Float16* __restrict__ pW2, const float* __restrict__ b2,
                         float* __restrict__ agg) {
  __shared__ __align__(16) _Float16 sm[EDGE_WAVES * 2 * 16 * kMsgKp];
  int wave = threadIdx.x >> 5;
  int lane = threadIdx.x & 31;
  int t0   = (blockIdx.x * EDGE_WAVES + wave) * TPW;
  _Float16* base = sm + wave * 2 * 16 * kMsgKp;
  const int* src = ei;
  const int* dst = ei + kE;
  int n = lane & 15, mb = (lane >> 4) * 8;

  edge_gather_async(base, t0, h16, ea16, d2p, src, dst);

  for (int it = 0; it < TPW; it++) {
    _Float16* A = base + (it & 1) * (16 * kMsgKp);
    int tile = t0 + it;
    if (it + 1 < TPW) {
      edge_gather_async(base + ((it + 1) & 1) * (16 * kMsgKp), tile + 1,
                        h16, ea16, d2p, src, dst);
      // newest batch (28 ops) may remain in flight; older batch must be done
      asm volatile("s_wait_asynccnt 28" ::: "memory");
    } else {
      async_fence_all();
    }

    int dstv[8];
#pragma unroll
    for (int v = 0; v < 8; v++) dstv[v] = dst[tile * 16 + mb + v];

    // stage 1: [16 x 224] @ [224 x 96]
    v8f acc[kCT];
#pragma unroll
    for (int ct = 0; ct < kCT; ct++) acc[ct] = zero8();
#pragma unroll
    for (int ks = 0; ks < 7; ks++) {
      v16h a = load_afrag(A, kMsgKp, ks);
#pragma unroll
      for (int ct = 0; ct < kCT; ct++)
        acc[ct] = wmma16(a, load_bfrag(pW1, 7, ct, ks), acc[ct]);
    }
    // silu -> hidden tile (reuse A region, stride 96)
#pragma unroll
    for (int ct = 0; ct < kCT; ct++) {
      float bias = b1[ct * 16 + n];
#pragma unroll
      for (int v = 0; v < 8; v++)
        A[(mb + v) * kH + ct * 16 + n] = (_Float16)silu(acc[ct][v] + bias);
    }
    lds_fence();

    // stage 2: [16 x 96] @ [96 x 96] -> silu -> atomic scatter by dst
#pragma unroll
    for (int ct = 0; ct < kCT; ct++) {
      v8f u = zero8();
#pragma unroll
      for (int ks = 0; ks < 3; ks++)
        u = wmma16(load_afrag(A, kH, ks), load_bfrag(pW2, 3, ct, ks), u);
      float bias = b2[ct * 16 + n];
#pragma unroll
      for (int v = 0; v < 8; v++) {
        float m = silu(u[v] + bias);
        atomicAdd(&agg[(size_t)dstv[v] * kH + ct * 16 + n], m);
      }
    }
  }
}

// ---------------- node MLP: h += silu([h|agg]@W1+b1)@W2+b2 ----------------

#define NODE_WAVES 4
__global__ void node_mlp(const float* __restrict__ agg,
                         const _Float16* __restrict__ pW1, const float* __restrict__ b1,
                         const _Float16* __restrict__ pW2, const float* __restrict__ b2,
                         float* __restrict__ h32, _Float16* __restrict__ h16) {
  __shared__ __align__(16) _Float16 sm[NODE_WAVES * 16 * kNodeIn];
  int wave = threadIdx.x >> 5;
  int lane = threadIdx.x & 31;
  int tile = blockIdx.x * NODE_WAVES + wave;
  if (tile >= kNodeTiles) return;
  _Float16* A = sm + wave * 16 * kNodeIn;
  {
    int row = lane & 15;
    int node = tile * 16 + row;
    if (lane < 16) {
      const _Float16* hp = h16 + (size_t)node * kH;
      _Float16* dp = A + row * kNodeIn;
#pragma unroll
      for (int i = 0; i < 12; i++) async_b128(dp + i * 8, hp + i * 8);
    } else {
      const float* ap = agg + (size_t)node * kH;
      _Float16* dp = A + row * kNodeIn + kH;
#pragma unroll
      for (int i = 0; i < kH; i++) dp[i] = (_Float16)ap[i];
    }
  }
  async_fence_all();
  lds_fence();

  int n = lane & 15, mb = (lane >> 4) * 8;
  v8f acc[kCT];
#pragma unroll
  for (int ct = 0; ct < kCT; ct++) acc[ct] = zero8();
#pragma unroll
  for (int ks = 0; ks < 6; ks++) {
    v16h a = load_afrag(A, kNodeIn, ks);
#pragma unroll
    for (int ct = 0; ct < kCT; ct++)
      acc[ct] = wmma16(a, load_bfrag(pW1, 6, ct, ks), acc[ct]);
  }
#pragma unroll
  for (int ct = 0; ct < kCT; ct++) {
    float bias = b1[ct * 16 + n];
#pragma unroll
    for (int v = 0; v < 8; v++)
      A[(mb + v) * kH + ct * 16 + n] = (_Float16)silu(acc[ct][v] + bias);
  }
  lds_fence();
#pragma unroll
  for (int ct = 0; ct < kCT; ct++) {
    v8f u = zero8();
#pragma unroll
    for (int ks = 0; ks < 3; ks++)
      u = wmma16(load_afrag(A, kH, ks), load_bfrag(pW2, 3, ct, ks), u);
    float bias = b2[ct * 16 + n];
#pragma unroll
    for (int v = 0; v < 8; v++) {
      size_t o = (size_t)(tile * 16 + mb + v) * kH + ct * 16 + n;
      float val = h32[o] + u[v] + bias;  // skip connection
      h32[o] = val;
      h16[o] = (_Float16)val;
    }
  }
}

// ---------------- pooling + head ----------------

__global__ void pool_kernel(const float* __restrict__ h32, const int* __restrict__ bidx,
                            float* __restrict__ pooled, float* __restrict__ counts) {
  int idx = blockIdx.x * blockDim.x + threadIdx.x;
  if (idx >= kN * kH) return;
  int node = idx / kH, c = idx % kH;
  int b = bidx[node];
  atomicAdd(&pooled[(size_t)b * kH + c], h32[idx]);
  if (c == 0) atomicAdd(&counts[b], 1.0f);
}

__global__ void final_kernel(const float* __restrict__ pooled, const float* __restrict__ counts,
                             const float* __restrict__ Wl, const float* __restrict__ bl,
                             float* __restrict__ out) {
  int b = threadIdx.x;
  if (b >= kB) return;
  float cnt = fmaxf(counts[b], 1.0f);
  float s = 0.f;
#pragma unroll
  for (int c = 0; c < kH; c++) s += (pooled[(size_t)b * kH + c] / cnt) * Wl[c];
  out[b] = s + bl[0];
}

// ---------------- host launcher ----------------

extern "C" void kernel_launch(void* const* d_in, const int* in_sizes, int n_in,
                              void* d_out, int out_size, void* d_ws, size_t ws_size,
                              hipStream_t stream) {
  (void)in_sizes; (void)n_in; (void)out_size; (void)ws_size;
  const float* x    = (const float*)d_in[0];
  const float* pos  = (const float*)d_in[1];
  const int*   ei   = (const int*)d_in[2];
  const float* ea   = (const float*)d_in[3];
  const int*   bidx = (const int*)d_in[4];
  const float* We   = (const float*)d_in[5];
  const float* be   = (const float*)d_in[6];
  const float* eW1  = (const float*)d_in[7];
  const float* eb1  = (const float*)d_in[8];
  const float* eW2  = (const float*)d_in[9];
  const float* eb2  = (const float*)d_in[10];
  const float* nW1  = (const float*)d_in[11];
  const float* nb1  = (const float*)d_in[12];
  const float* nW2  = (const float*)d_in[13];
  const float* nb2  = (const float*)d_in[14];
  const float* Wl   = (const float*)d_in[15];
  const float* bl   = (const float*)d_in[16];
  float* out = (float*)d_out;

  char* ws = (char*)d_ws;
  size_t off = 0;
  auto alloc = [&](size_t bytes) -> void* {
    void* p = ws + off;
    off = (off + bytes + 255) & ~(size_t)255;
    return p;
  };
  float*    h32  = (float*)alloc((size_t)kN * kH * 4);
  _Float16* h16  = (_Float16*)alloc((size_t)kN * kH * 2);
  float*    agg  = (float*)alloc((size_t)kN * kH * 4);
  _Float16* ea16 = (_Float16*)alloc((size_t)kE * kEdgeDim * 2);
  _Float16* d2p  = (_Float16*)alloc((size_t)kE * 16 * 2);
  _Float16* pWe  = (_Float16*)alloc((size_t)kCT * 1 * 512 * 2);
  _Float16 *pE1[3], *pE2[3], *pN1[3], *pN2[3];
  for (int l = 0; l < 3; l++) {
    pE1[l] = (_Float16*)alloc((size_t)kCT * 7 * 512 * 2);
    pE2[l] = (_Float16*)alloc((size_t)kCT * 3 * 512 * 2);
    pN1[l] = (_Float16*)alloc((size_t)kCT * 6 * 512 * 2);
    pN2[l] = (_Float16*)alloc((size_t)kCT * 3 * 512 * 2);
  }
  float* pooled = (float*)alloc((size_t)(kB * kH + kB) * 4);
  float* counts = pooled + (size_t)kB * kH;

  auto launch_pack = [&](const float* W, int K, int numK, int mode, _Float16* dst) {
    int total = kCT * numK * 512;
    pack_weights<<<(total + 255) / 256, 256, 0, stream>>>(W, K, numK, mode, dst);
  };
  launch_pack(We, kFIn, 1, 0, pWe);
  for (int l = 0; l < 3; l++) {
    launch_pack(eW1 + (size_t)l * kMsgIn * kH, kMsgIn, 7, 1, pE1[l]);
    launch_pack(eW2 + (size_t)l * kH * kH,     kH,     3, 0, pE2[l]);
    launch_pack(nW1 + (size_t)l * kNodeIn * kH, kNodeIn, 6, 0, pN1[l]);
    launch_pack(nW2 + (size_t)l * kH * kH,     kH,     3, 0, pN2[l]);
  }

  edge_pre<<<(kE + 255) / 256, 256, 0, stream>>>(ei, pos, ea, ea16, d2p);
  encoder_kernel<<<(kNodeTiles + ENC_WAVES - 1) / ENC_WAVES, ENC_WAVES * 32, 0, stream>>>(
      x, pWe, be, h32, h16);

  for (int l = 0; l < 3; l++) {
    zero_kernel<<<(kN * kH + 255) / 256, 256, 0, stream>>>(agg, kN * kH);
    edge_mlp<<<kEdgeTiles / (EDGE_WAVES * TPW), EDGE_WAVES * 32, 0, stream>>>(
        h16, ea16, d2p, ei, pE1[l], eb1 + l * kH, pE2[l], eb2 + l * kH, agg);
    node_mlp<<<(kNodeTiles + NODE_WAVES - 1) / NODE_WAVES, NODE_WAVES * 32, 0, stream>>>(
        agg, pN1[l], nb1 + l * kH, pN2[l], nb2 + l * kH, h32, h16);
  }

  zero_kernel<<<(kB * kH + kB + 255) / 256, 256, 0, stream>>>(pooled, kB * kH + kB);
  pool_kernel<<<(kN * kH + 255) / 256, 256, 0, stream>>>(h32, bidx, pooled, counts);
  final_kernel<<<1, 64, 0, stream>>>(pooled, counts, Wl, bl, out);
}
